// ResidualBlock_78503412236511
// MI455X (gfx1250) — matrix operand order repaired and checked by reading, not compiled, not run on hardware
//
#include <hip/hip_runtime.h>

// ---------------------------------------------------------------------------
// Problem constants (from reference)
// ---------------------------------------------------------------------------
#define NPTS   400000
#define INC    128
#define MIDC   64
#define OUTC   128
#define KVOL   27
#define BN_EPS 1e-5f

// Tiles of 16 rows, one 16-row tile per wave.
#define NTILES (NPTS / 16)          // 25000
#define K1_BLOCKS (NTILES / 8)      // 3125 (8 waves/block)
#define K2_BLOCKS 625               // 625*8 = 5000 wave-slots -> 5 tiles each

typedef __attribute__((ext_vector_type(16))) __bf16 v16bf;
typedef __attribute__((ext_vector_type(8)))  float  v8f;

// ---------------------------------------------------------------------------
// helpers
// ---------------------------------------------------------------------------
__device__ __forceinline__ unsigned short f2bf(float f) {
  union { float f; unsigned u; } c; c.f = f;
  unsigned u = c.u;
  return (unsigned short)((u + 0x7FFFu + ((u >> 16) & 1u)) >> 16);
}

// Packed f32x2 -> bf16x2 (one VALU op when the HW builtin is available).
__device__ __forceinline__ unsigned f2bf_pk(float lo, float hi) {
#if __has_builtin(__builtin_amdgcn_cvt_pk_bf16_f32)
  typedef __attribute__((ext_vector_type(2))) __bf16 v2bf;
  union { v2bf v; unsigned u; } c;
  c.v = __builtin_amdgcn_cvt_pk_bf16_f32(lo, hi);
  return c.u;
#else
  return (unsigned)f2bf(lo) | ((unsigned)f2bf(hi) << 16);
#endif
}

union AB16 { v16bf v; uint4 q[2]; unsigned u32[8]; unsigned short us[16]; };

// Load a 16-element bf16 fragment: elements e0..15 contiguous at base+off
// (contiguous k order k = 16*(lane>>4) + e, matching our packed B-fragment
// layout where each lane's 16 values are contiguous: off = frag*512 + lane*16).
__device__ __forceinline__ v16bf load_frag_bf16(const unsigned short* base, int off) {
  AB16 u;
  u.q[0] = *reinterpret_cast<const uint4*>(base + off);
  u.q[1] = *reinterpret_cast<const uint4*>(base + off + 8);
  return u.v;
}

// Build an A fragment (16x32 bf16) from a row-major f32 row.
// cbase = kt*32 + 16*(lane>=16): cols [cbase,cbase+16) contiguous.
__device__ __forceinline__ v16bf load_a_f32(const float* row, int cbase) {
  AB16 u;
  const float4 a = *reinterpret_cast<const float4*>(row + cbase);
  const float4 b = *reinterpret_cast<const float4*>(row + cbase + 4);
  const float4 c = *reinterpret_cast<const float4*>(row + cbase + 8);
  const float4 d = *reinterpret_cast<const float4*>(row + cbase + 12);
  u.u32[0] = f2bf_pk(a.x, a.y);
  u.u32[1] = f2bf_pk(a.z, a.w);
  u.u32[2] = f2bf_pk(b.x, b.y);
  u.u32[3] = f2bf_pk(b.z, b.w);
  u.u32[4] = f2bf_pk(c.x, c.y);
  u.u32[5] = f2bf_pk(c.z, c.w);
  u.u32[6] = f2bf_pk(d.x, d.y);
  u.u32[7] = f2bf_pk(d.z, d.w);
  return u.v;
}

__device__ __forceinline__ v8f wmma_bf16(v16bf a, v16bf b, v8f c) {
  return __builtin_amdgcn_wmma_f32_16x16x32_bf16(
      /*neg_a=*/false, a, /*neg_b=*/false, b,
      /*c_mod=*/(short)0, c, /*reuse_a=*/false, /*reuse_b=*/false);
}

// Pack one element of a B fragment. W is [K x Nc] row-major f32.
// Fragment f covers K rows [32*kt, 32*kt+32), cols [16*nt, 16*nt+16).
// Per-lane layout: lane L holds column (L&15), K = 32*kt + 16*(L>>4) + e.
__device__ __forceinline__ void pack_frag_elem(const float* W, int Nc, int NT,
                                               int i, unsigned short* dst) {
  int f  = i >> 9;
  int L  = (i >> 4) & 31;
  int e  = i & 15;
  int kt = f / NT, nt = f % NT;
  int K  = 32 * kt + 16 * (L >> 4) + e;
  int c  = 16 * nt + (L & 15);
  dst[i] = f2bf(W[K * Nc + c]);
}

// ---------------------------------------------------------------------------
// Kernel 0: fold BN params, convert + swizzle weights to bf16 fragments
// ---------------------------------------------------------------------------
__global__ __launch_bounds__(256)
void pack_kernel(const float* __restrict__ w1, const float* __restrict__ w2,
                 const float* __restrict__ w3,
                 const float* g1, const float* b1, const float* m1, const float* v1,
                 const float* g2, const float* b2, const float* m2, const float* v2,
                 const float* g3, const float* b3, const float* m3, const float* v3,
                 unsigned short* w1p, unsigned short* w2p, unsigned short* w3p,
                 float* s1, float* t1, float* s2, float* t2, float* s3, float* t3) {
  int tid = blockIdx.x * blockDim.x + threadIdx.x;
  int nth = gridDim.x * blockDim.x;

  // w1: 128x64 -> 4 kt * 4 nt = 16 frags
  for (int i = tid; i < 16 * 512; i += nth) pack_frag_elem(w1, 64, 4, i, w1p);
  // w2: 27 matrices of 64x64 -> per-k 2 kt * 4 nt = 8 frags (4096 elems)
  for (int i = tid; i < KVOL * 4096; i += nth) {
    int k = i >> 12, rem = i & 4095;
    pack_frag_elem(w2 + k * 4096, 64, 4, rem, w2p + (k << 12));
  }
  // w3: 64x128 -> 2 kt * 8 nt = 16 frags
  for (int i = tid; i < 16 * 512; i += nth) pack_frag_elem(w3, 128, 8, i, w3p);

  // BN folds: y = h*s + t ; s = g*rsqrt(v+eps), t = b - m*s
  for (int i = tid; i < MIDC; i += nth) {
    float s = g1[i] * rsqrtf(v1[i] + BN_EPS);
    s1[i] = s; t1[i] = b1[i] - m1[i] * s;
  }
  for (int i = tid; i < MIDC; i += nth) {
    float s = g2[i] * rsqrtf(v2[i] + BN_EPS);
    s2[i] = s; t2[i] = b2[i] - m2[i] * s;
  }
  for (int i = tid; i < OUTC; i += nth) {
    float s = g3[i] * rsqrtf(v3[i] + BN_EPS);
    s3[i] = s; t3[i] = b3[i] - m3[i] * s;
  }
}

// ---------------------------------------------------------------------------
// Kernel 1: h = relu(bn1(x @ w1)) -> bf16 [N,64]  (fits in L2)
// ---------------------------------------------------------------------------
__global__ __launch_bounds__(256)
void k1_in_gemm(const float* __restrict__ x, const unsigned short* __restrict__ w1p,
                const float* __restrict__ s1, const float* __restrict__ t1,
                unsigned short* __restrict__ hbf) {
  __shared__ unsigned short w1l[16 * 512];   // 16 KB of packed w1 fragments
  {
    const uint4* src = reinterpret_cast<const uint4*>(w1p);
    uint4* dst = reinterpret_cast<uint4*>(w1l);
    for (int i = threadIdx.x; i < 1024; i += 256) dst[i] = src[i];
  }
  __syncthreads();

  const int wave = threadIdx.x >> 5;
  const int lane = threadIdx.x & 31;
  const int l15  = lane & 15;
  const int hl   = lane >> 4;

  const int tile = blockIdx.x * 8 + wave;
  const int m0   = tile * 16;

  const float* arow = x + (size_t)(m0 + l15) * INC;

  const v8f vzero = {0.f,0.f,0.f,0.f,0.f,0.f,0.f,0.f};
  v8f acc[4] = {vzero, vzero, vzero, vzero};

  float s1v[4], t1v[4];
#pragma unroll
  for (int nt = 0; nt < 4; ++nt) {
    int c = nt * 16 + l15;
    s1v[nt] = s1[c]; t1v[nt] = t1[c];
  }

#pragma unroll
  for (int kt = 0; kt < 4; ++kt) {
    v16bf a = load_a_f32(arow, kt * 32 + 16 * hl);
#pragma unroll
    for (int nt = 0; nt < 4; ++nt) {
      v16bf b = load_frag_bf16(w1l, (kt * 4 + nt) * 512 + lane * 16);
      acc[nt] = wmma_bf16(a, b, acc[nt]);
    }
  }

  // bn1 + relu -> bf16 store (row-major [N,64])
#pragma unroll
  for (int nt = 0; nt < 4; ++nt) {
#pragma unroll
    for (int r = 0; r < 8; ++r) {
      float v = acc[nt][r] * s1v[nt] + t1v[nt];
      v = fmaxf(v, 0.0f);
      size_t row = (size_t)(m0 + r + 8 * hl);
      hbf[row * MIDC + nt * 16 + l15] = f2bf(v);
    }
  }
}

// ---------------------------------------------------------------------------
// Kernel 2 (fused): h2 = Σ_k h[nb[k]] @ w2[k]; relu(bn2(h2));
//                   out = relu(bn3(h2 @ w3) + x)
// LDS: 216KB w2 frags + 16KB w3 frags + 16KB staging = 248KB (one WG / WGP)
// ---------------------------------------------------------------------------
#define W2L_ELEMS (KVOL * 8 * 512)          // 110592 ushort
#define W3L_ELEMS (16 * 512)                // 8192 ushort
#define STG_ELEMS (8 * 1024)                // 8 waves * 16x64 bf16
#define SMEM_K2_BYTES ((W2L_ELEMS + W3L_ELEMS + STG_ELEMS) * 2)  // 253952

__global__ __launch_bounds__(256)
void k2_fused(const unsigned short* __restrict__ hbf, const int* __restrict__ nb,
              const unsigned short* __restrict__ w2p,
              const unsigned short* __restrict__ w3p,
              const float* __restrict__ s2, const float* __restrict__ t2,
              const float* __restrict__ s3, const float* __restrict__ t3,
              const float* __restrict__ x, float* __restrict__ out) {
  extern __shared__ unsigned short sm[];
  unsigned short* w2l = sm;
  unsigned short* w3l = sm + W2L_ELEMS;
  unsigned short* stg = sm + W2L_ELEMS + W3L_ELEMS;

  {
    const uint4* s2g = reinterpret_cast<const uint4*>(w2p);
    uint4* d2 = reinterpret_cast<uint4*>(w2l);
    for (int i = threadIdx.x; i < W2L_ELEMS / 8; i += 256) d2[i] = s2g[i];
    const uint4* s3g = reinterpret_cast<const uint4*>(w3p);
    uint4* d3 = reinterpret_cast<uint4*>(w3l);
    for (int i = threadIdx.x; i < W3L_ELEMS / 8; i += 256) d3[i] = s3g[i];
  }
  __syncthreads();

  const int wave = threadIdx.x >> 5;
  const int lane = threadIdx.x & 31;
  const int l15  = lane & 15;
  const int hl   = lane >> 4;

  unsigned short* mystg = stg + wave * 1024;  // this wave's 16x64 bf16 tile

  float s2v[4], t2v[4], s3v[8], t3v[8];
#pragma unroll
  for (int nt = 0; nt < 4; ++nt) {
    int c = nt * 16 + l15; s2v[nt] = s2[c]; t2v[nt] = t2[c];
  }
#pragma unroll
  for (int nt = 0; nt < 8; ++nt) {
    int c = nt * 16 + l15; s3v[nt] = s3[c]; t3v[nt] = t3[c];
  }

  const v8f vzero = {0.f,0.f,0.f,0.f,0.f,0.f,0.f,0.f};

  for (int tile = blockIdx.x * 8 + wave; tile < NTILES; tile += gridDim.x * 8) {
    const int m0 = tile * 16;

    // ---- 27-tap gather GEMM: acc2 = Σ_k h[nb[k][rows]] @ w2[k] ----
    v8f acc2[4] = {vzero, vzero, vzero, vzero};
    for (int k = 0; k < KVOL; ++k) {
      int idx = nb[k * NPTS + m0 + l15];           // per-lane gathered row
      const unsigned short* hrow = hbf + (size_t)idx * MIDC;
#pragma unroll
      for (int kt = 0; kt < 2; ++kt) {
        v16bf a = load_frag_bf16(hrow, kt * 32 + 16 * hl);
#pragma unroll
        for (int nt = 0; nt < 4; ++nt) {
          v16bf b = load_frag_bf16(w2l, ((k * 2 + kt) * 4 + nt) * 512 + lane * 16);
          acc2[nt] = wmma_bf16(a, b, acc2[nt]);
        }
      }
    }

    // ---- bn2 + relu -> bf16 staging tile (LDS, wave-private) ----
#pragma unroll
    for (int nt = 0; nt < 4; ++nt) {
#pragma unroll
      for (int r = 0; r < 8; ++r) {
        float v = acc2[nt][r] * s2v[nt] + t2v[nt];
        v = fmaxf(v, 0.0f);
        mystg[(r + 8 * hl) * MIDC + nt * 16 + l15] = f2bf(v);
      }
    }
    // DS ops are wave-in-order; explicit wait for belt-and-braces.
    asm volatile("s_wait_dscnt 0" ::: "memory");

    // ---- output GEMM: acc3 = stage @ w3 (64 -> 128) ----
    v8f acc3[8] = {vzero, vzero, vzero, vzero, vzero, vzero, vzero, vzero};
#pragma unroll
    for (int kt = 0; kt < 2; ++kt) {
      v16bf a = load_frag_bf16(mystg + l15 * MIDC, kt * 32 + 16 * hl);
#pragma unroll
      for (int nt = 0; nt < 8; ++nt) {
        v16bf b = load_frag_bf16(w3l, (kt * 8 + nt) * 512 + lane * 16);
        acc3[nt] = wmma_bf16(a, b, acc3[nt]);
      }
    }

    // ---- bn3 + residual + relu -> f32 out ----
#pragma unroll
    for (int nt = 0; nt < 8; ++nt) {
#pragma unroll
      for (int r = 0; r < 8; ++r) {
        size_t row = (size_t)(m0 + r + 8 * hl);
        int col = nt * 16 + l15;
        float v = acc3[nt][r] * s3v[nt] + t3v[nt] + x[row * OUTC + col];
        out[row * OUTC + col] = fmaxf(v, 0.0f);
      }
    }
  }
}

// ---------------------------------------------------------------------------
// launch
// ---------------------------------------------------------------------------
extern "C" void kernel_launch(void* const* d_in, const int* in_sizes, int n_in,
                              void* d_out, int out_size, void* d_ws, size_t ws_size,
                              hipStream_t stream) {
  (void)in_sizes; (void)n_in; (void)out_size; (void)ws_size;

  const float* x  = (const float*)d_in[0];
  const int*   nb = (const int*)  d_in[1];
  const float* w1 = (const float*)d_in[2];
  const float* w2 = (const float*)d_in[3];
  const float* w3 = (const float*)d_in[4];
  const float* g1 = (const float*)d_in[5];
  const float* b1 = (const float*)d_in[6];
  const float* m1 = (const float*)d_in[7];
  const float* v1 = (const float*)d_in[8];
  const float* g2 = (const float*)d_in[9];
  const float* b2 = (const float*)d_in[10];
  const float* m2 = (const float*)d_in[11];
  const float* v2 = (const float*)d_in[12];
  const float* g3 = (const float*)d_in[13];
  const float* b3 = (const float*)d_in[14];
  const float* m3 = (const float*)d_in[15];
  const float* v3 = (const float*)d_in[16];
  float* out = (float*)d_out;

  // workspace layout (all 256B aligned)
  char* ws = (char*)d_ws;
  unsigned short* hbf = (unsigned short*)ws;                       // 51,200,000 B
  size_t off = (size_t)NPTS * MIDC * 2;
  unsigned short* w1p = (unsigned short*)(ws + off); off += 16 * 512 * 2;
  unsigned short* w2p = (unsigned short*)(ws + off); off += KVOL * 4096 * 2;
  unsigned short* w3p = (unsigned short*)(ws + off); off += 16 * 512 * 2;
  float* s1 = (float*)(ws + off); off += MIDC * 4;
  float* t1 = (float*)(ws + off); off += MIDC * 4;
  float* s2 = (float*)(ws + off); off += MIDC * 4;
  float* t2 = (float*)(ws + off); off += MIDC * 4;
  float* s3 = (float*)(ws + off); off += OUTC * 4;
  float* t3 = (float*)(ws + off);

  (void)hipFuncSetAttribute(reinterpret_cast<const void*>(k2_fused),
                            hipFuncAttributeMaxDynamicSharedMemorySize,
                            (int)SMEM_K2_BYTES);

  pack_kernel<<<64, 256, 0, stream>>>(w1, w2, w3,
                                      g1, b1, m1, v1,
                                      g2, b2, m2, v2,
                                      g3, b3, m3, v3,
                                      w1p, w2p, w3p,
                                      s1, t1, s2, t2, s3, t3);

  k1_in_gemm<<<K1_BLOCKS, 256, 0, stream>>>(x, w1p, s1, t1, hbf);

  k2_fused<<<K2_BLOCKS, 256, SMEM_K2_BYTES, stream>>>(hbf, nb, w2p, w3p,
                                                      s2, t2, s3, t3, x, out);
}
